// AdaptiveSoftmax_50491635531945
// MI455X (gfx1250) — compile-verified, matrix operand order checked by reading, not run on hardware
//
#include <hip/hip_runtime.h>
#include <hip/hip_bf16.h>
#include <stdint.h>

// ---------------------------------------------------------------------------
// Adaptive softmax (head 20002 + two tails) fused bf16-WMMA implementation
// for gfx1250 (MI455X).  GEMMs run on v_wmma_f32_16x16x32_bf16; CE is fused
// (online softmax, logits never materialized); A-panels staged into LDS via
// the Tensor Data Mover (tensor_load_to_lds + s_wait_tensorcnt).
// ---------------------------------------------------------------------------

typedef __attribute__((ext_vector_type(16))) __bf16 v16bf;
typedef __attribute__((ext_vector_type(8)))  float  v8f;
typedef __attribute__((ext_vector_type(4)))  unsigned int v4u;
typedef __attribute__((ext_vector_type(8)))  int v8i;
typedef __attribute__((ext_vector_type(4)))  int v4i;

#define N_ROWS 4096
#define HID    1024
#define NEG_BIG (-3.0e38f)

static __device__ __forceinline__ unsigned short bf_bits(float f) {
  __bf16 b = (__bf16)f;
  union { __bf16 b; unsigned short u; } cv; cv.b = b; return cv.u;
}

// ---------------------------------------------------------------------------
// Convert x (fp32 [N,H]) to bf16 in WMMA A-fragment-swizzled layout:
//   offset(halves) = ((mtile*CH + kchunk)*32 + lane)*16 + h
//   lane = (m&15) | (((k>>3)&1)<<4),  h = ((k&16)>>1) | (k&7)
// Each thread handles an even (k,k+1) pair -> one 4-byte store.
// ---------------------------------------------------------------------------
__global__ void k_convert_x(const float* __restrict__ x,
                            unsigned int* __restrict__ xb) {
  int p = blockIdx.x * blockDim.x + threadIdx.x;
  int total = N_ROWS * (HID / 2);
  if (p >= total) return;
  int row = p / (HID / 2);
  int k   = (p - row * (HID / 2)) * 2;
  float f0 = x[(size_t)row * HID + k];
  float f1 = x[(size_t)row * HID + k + 1];
  int mt = row >> 4, kc = k >> 5;
  int lane = (row & 15) | (((k >> 3) & 1) << 4);
  int h = ((k & 16) >> 1) | (k & 7);
  size_t off = (((size_t)(mt * (HID >> 5) + kc)) * 32 + lane) * 16 + h;
  xb[off >> 1] = (unsigned)bf_bits(f0) | ((unsigned)bf_bits(f1) << 16);
}

// ---------------------------------------------------------------------------
// Convert weight (fp32 [K,V] row-major) to bf16 B-fragment-swizzled layout
// (lane = column within 16-tile, same h mapping).  Columns >= V -> 0.
// ---------------------------------------------------------------------------
__global__ void k_convert_w(const float* __restrict__ w,
                            unsigned int* __restrict__ wb,
                            int K, int V, int Vp) {
  long p = (long)blockIdx.x * blockDim.x + threadIdx.x;
  long total = (long)Vp * (K / 2);
  if (p >= total) return;
  int col = (int)(p % Vp);
  int k   = (int)(p / Vp) * 2;
  float f0 = 0.f, f1 = 0.f;
  if (col < V) {
    f0 = w[(size_t)k * V + col];
    f1 = w[(size_t)(k + 1) * V + col];
  }
  int CH = K >> 5;
  int ct = col >> 4, kc = k >> 5;
  int lane = (col & 15) | (((k >> 3) & 1) << 4);
  int h = ((k & 16) >> 1) | (k & 7);
  size_t off = (((size_t)(ct * CH + kc)) * 32 + lane) * 16 + h;
  wb[off >> 1] = (unsigned)bf_bits(f0) | ((unsigned)bf_bits(f1) << 16);
}

// ---------------------------------------------------------------------------
// Tail projection: H = xb @ pw + pb, output written bf16 pre-swizzled as the
// A operand of the subsequent tail GEMM.  One wave per 16x16 output tile.
// ---------------------------------------------------------------------------
__global__ void k_proj(const v16bf* __restrict__ xb,
                       const v16bf* __restrict__ pwb,
                       const float* __restrict__ pb,
                       __bf16* __restrict__ hb,
                       int CH, int CH2) {
  int mt = blockIdx.x, ct = blockIdx.y, lane = threadIdx.x;
  v8f c = {0.f, 0.f, 0.f, 0.f, 0.f, 0.f, 0.f, 0.f};
  for (int kc = 0; kc < CH; ++kc) {
    v16bf a = xb[(size_t)(mt * CH + kc) * 32 + lane];
    v16bf b = pwb[(size_t)(ct * CH + kc) * 32 + lane];
    c = __builtin_amdgcn_wmma_f32_16x16x32_bf16(false, a, false, b,
                                                (short)0, c, false, false);
  }
  int n15 = lane & 15, g = lane >> 4;
  float bias = pb[ct * 16 + n15];
  int ko = ct * 16 + n15;           // output col becomes k-index of next GEMM
  int lhi = ((ko >> 3) & 1) << 4;
  int h2 = ((ko & 16) >> 1) | (ko & 7);
  int kc2 = ko >> 5;
#pragma unroll
  for (int r = 0; r < 8; ++r) {
    int m = mt * 16 + r + 8 * g;
    size_t off = (((size_t)(mt * CH2 + kc2)) * 32 + ((m & 15) | lhi)) * 16 + h2;
    hb[off] = (__bf16)(c[r] + bias);
  }
}

// ---------------------------------------------------------------------------
// Fused GEMM + online-softmax cross-entropy for one cluster.
// 64 rows/block, 8 waves = 4 M-tiles x 2 vocab halves.  A panel is DMA'd
// into LDS by the TDM; B fragments stream from global (weights live in L2).
// mode 0 = head (labels remapped to cluster tokens), mode 1/2 = tails.
// ---------------------------------------------------------------------------
__global__ void k_ce(const __bf16* __restrict__ ab,
                     const v16bf* __restrict__ wb,
                     const float* __restrict__ bias,
                     const long long* __restrict__ labels,
                     float* __restrict__ out,
                     int K, int CH, int V, int VT,
                     int mode, int lo, int hi) {
  extern __shared__ char smem[];
  const int rows = 64;
  int rowBase = blockIdx.x * rows;
  int tid = threadIdx.x;
  int wave = tid >> 5, lane = tid & 31;
  int mt = wave & 3, colHalf = wave >> 2;
  int g = lane >> 4, n15 = lane & 15;
  size_t panelBytes = (size_t)rows * K * 2;
  float* pm = (float*)(smem + panelBytes);
  float* ps = pm + rows * 2;
  float* pl = ps + rows * 2;

  // ---- TDM: async copy of the contiguous swizzled A panel into LDS ----
  if (tid < 32) {
    unsigned long long ga =
        (unsigned long long)(uintptr_t)(ab + (size_t)rowBase * K);
    unsigned ldsOff = (unsigned)(uintptr_t)smem;
    unsigned el = (unsigned)(panelBytes >> 3);   // 8-byte elements (<=16384)
    v4u g0 = { 1u,                                // count=1, user mode
               ldsOff,                            // lds_addr
               (unsigned)(ga & 0xffffffffu),      // global_addr[31:0]
               (unsigned)((ga >> 32) & 0x1ffffffu) | (2u << 30) }; // type=2
    v8i g1 = { (int)(3u << 16),                   // data_size = 8B
               (int)((el & 0xffffu) << 16),       // tensor_dim0[15:0]
               (int)(((el >> 16) & 0xffffu) | (1u << 16)), // dim0 hi, dim1=1
               (int)((el & 0xffffu) << 16),       // tile_dim0
               1,                                 // tile_dim1 = 1
               (int)el,                           // tensor_dim0_stride lo
               0, 0 };
    v4i g2 = {0, 0, 0, 0}, g3 = {0, 0, 0, 0};
    v8i g4 = {0, 0, 0, 0, 0, 0, 0, 0};
    __builtin_amdgcn_tensor_load_to_lds(g0, g1, g2, g3, g4, 0);
    __builtin_amdgcn_s_wait_tensorcnt(0);
  }
  __syncthreads();

  // ---- per-row effective labels for this wave's 16-row tile ----
  int effLab[8];
#pragma unroll
  for (int r = 0; r < 8; ++r) {
    long long ll = labels[rowBase + mt * 16 + r + 8 * g];
    if (mode == 0)
      effLab[r] = (ll >= 40000) ? 20001 : (ll >= 20000 ? 20000 : (int)ll);
    else
      effLab[r] = (ll >= lo && ll < hi) ? (int)(ll - lo) : -1;
  }
  float runM[8], runS[8], labA[8];
#pragma unroll
  for (int r = 0; r < 8; ++r) { runM[r] = NEG_BIG; runS[r] = 0.f; labA[r] = 0.f; }

  const v16bf* panel = (const v16bf*)smem;
  for (int ct = colHalf; ct < VT; ct += 2) {
    v8f c = {0.f, 0.f, 0.f, 0.f, 0.f, 0.f, 0.f, 0.f};
    for (int kc = 0; kc < CH; ++kc) {
      v16bf a = panel[(mt * CH + kc) * 32 + lane];
      v16bf b = wb[(size_t)(ct * CH + kc) * 32 + lane];
      c = __builtin_amdgcn_wmma_f32_16x16x32_bf16(false, a, false, b,
                                                  (short)0, c, false, false);
    }
    int col = ct * 16 + n15;
    bool cv = col < V;
    float bv = cv ? bias[col] : 0.f;
#pragma unroll
    for (int r = 0; r < 8; ++r) {
      float v = cv ? (c[r] + bv) : NEG_BIG;
      float lv = (col == effLab[r]) ? v : 0.f;
      float tmax = v;
      for (int msk = 1; msk < 16; msk <<= 1)
        tmax = fmaxf(tmax, __shfl_xor(tmax, msk, 16));
      float nm = fmaxf(runM[r], tmax);
      float e = __expf(v - nm);
      float es = e, ls = lv;
      for (int msk = 1; msk < 16; msk <<= 1) {
        es += __shfl_xor(es, msk, 16);
        ls += __shfl_xor(ls, msk, 16);
      }
      runS[r] = runS[r] * __expf(runM[r] - nm) + es;
      runM[r] = nm;
      labA[r] += ls;
    }
  }

  // ---- combine the two vocab halves per row ----
  if (n15 == 0) {
#pragma unroll
    for (int r = 0; r < 8; ++r) {
      int rl = mt * 16 + r + 8 * g;
      pm[rl * 2 + colHalf] = runM[r];
      ps[rl * 2 + colHalf] = runS[r];
      pl[rl * 2 + colHalf] = labA[r];
    }
  }
  __syncthreads();
  if (tid < rows) {
    float M0 = pm[tid * 2], M1 = pm[tid * 2 + 1];
    float Mc = fmaxf(M0, M1);
    float Sc = ps[tid * 2] * __expf(M0 - Mc) + ps[tid * 2 + 1] * __expf(M1 - Mc);
    float L = pl[tid * 2] + pl[tid * 2 + 1];
    float loss = __logf(Sc) + Mc - L;       // logsumexp - label logit
    long long ll = labels[rowBase + tid];
    if (mode != 0 && !(ll >= lo && ll < hi)) loss = 0.f;
    out[rowBase + tid] = loss;
  }
}

// ---------------------------------------------------------------------------
// Final reduction: mean over rows of (head + tail1 + tail2) * pad_mask.
// ---------------------------------------------------------------------------
__global__ void k_reduce(const float* __restrict__ a,
                         const float* __restrict__ b,
                         const float* __restrict__ c,
                         const long long* __restrict__ labels,
                         float* __restrict__ out) {
  __shared__ float red[256];
  int t = threadIdx.x;
  float acc = 0.f;
  for (int i = t; i < N_ROWS; i += 256) {
    float pad = (labels[i] != 0) ? 1.f : 0.f;
    acc += (a[i] + b[i] + c[i]) * pad;
  }
  red[t] = acc;
  __syncthreads();
  for (int s = 128; s > 0; s >>= 1) {
    if (t < s) red[t] += red[t + s];
    __syncthreads();
  }
  if (t == 0) out[0] = red[0] / (float)N_ROWS;
}

// ---------------------------------------------------------------------------
extern "C" void kernel_launch(void* const* d_in, const int* in_sizes, int n_in,
                              void* d_out, int out_size, void* d_ws,
                              size_t ws_size, hipStream_t stream) {
  (void)in_sizes; (void)n_in; (void)out_size; (void)ws_size;
  const float*     inp    = (const float*)d_in[0];
  const long long* labels = (const long long*)d_in[1];
  const float*     head_w = (const float*)d_in[2];
  const float*     head_b = (const float*)d_in[3];
  const float*     t1_pw  = (const float*)d_in[4];
  const float*     t1_pb  = (const float*)d_in[5];
  const float*     t1_ow  = (const float*)d_in[6];
  const float*     t1_ob  = (const float*)d_in[7];
  const float*     t2_pw  = (const float*)d_in[8];
  const float*     t2_pb  = (const float*)d_in[9];
  const float*     t2_ow  = (const float*)d_in[10];
  const float*     t2_ob  = (const float*)d_in[11];

  char* ws = (char*)d_ws;
  size_t off = 0;
  __bf16* xb  = (__bf16*)(ws + off); off += (size_t)4096 * 1024 * 2;   // x swizzled
  __bf16* h1b = (__bf16*)(ws + off); off += (size_t)4096 * 256 * 2;    // tail1 hidden
  __bf16* h2b = (__bf16*)(ws + off); off += (size_t)4096 * 64 * 2;     // tail2 hidden
  __bf16* whb = (__bf16*)(ws + off); off += (size_t)20016 * 1024 * 2;  // head W
  __bf16* w1b = (__bf16*)(ws + off); off += (size_t)20000 * 256 * 2;   // tail1 W
  __bf16* w2b = (__bf16*)(ws + off); off += (size_t)10000 * 64 * 2;    // tail2 W
  __bf16* p1b = (__bf16*)(ws + off); off += (size_t)256 * 1024 * 2;    // proj1 W
  __bf16* p2b = (__bf16*)(ws + off); off += (size_t)64 * 1024 * 2;     // proj2 W
  float* ceH = (float*)(ws + off); off += 4096 * 4;
  float* ce1 = (float*)(ws + off); off += 4096 * 4;
  float* ce2 = (float*)(ws + off); off += 4096 * 4;

  { // x -> bf16 swizzled
    int pairs = 4096 * 1024 / 2;
    k_convert_x<<<(pairs + 255) / 256, 256, 0, stream>>>(inp, (unsigned int*)xb);
  }
  auto cw = [&](const float* w, __bf16* wbp, int K, int V, int Vp) {
    long total = (long)Vp * (K / 2);
    k_convert_w<<<(unsigned)((total + 255) / 256), 256, 0, stream>>>(
        w, (unsigned int*)wbp, K, V, Vp);
  };
  cw(head_w, whb, 1024, 20002, 20016);
  cw(t1_ow,  w1b, 256, 20000, 20000);
  cw(t2_ow,  w2b, 64, 10000, 10000);
  cw(t1_pw,  p1b, 1024, 256, 256);
  cw(t2_pw,  p2b, 1024, 64, 64);

  // tail projections (WMMA)
  k_proj<<<dim3(256, 16), 32, 0, stream>>>((const v16bf*)xb, (const v16bf*)p1b,
                                           t1_pb, h1b, 32, 8);
  k_proj<<<dim3(256, 4), 32, 0, stream>>>((const v16bf*)xb, (const v16bf*)p2b,
                                          t2_pb, h2b, 32, 2);

  // fused GEMM + CE per cluster
  k_ce<<<64, 256, 64 * 1024 * 2 + 1536, stream>>>(
      xb, (const v16bf*)whb, head_b, labels, ceH, 1024, 32, 20002, 1251, 0, 0, 0);
  k_ce<<<64, 256, 64 * 256 * 2 + 1536, stream>>>(
      h1b, (const v16bf*)w1b, t1_ob, labels, ce1, 256, 8, 20000, 1250, 1, 20000, 40000);
  k_ce<<<64, 256, 64 * 64 * 2 + 1536, stream>>>(
      h2b, (const v16bf*)w2b, t2_ob, labels, ce2, 64, 2, 10000, 625, 2, 40000, 50000);

  k_reduce<<<1, 256, 0, stream>>>(ceH, ce1, ce2, labels, (float*)d_out);
}
